// NGP_44478681317674
// MI455X (gfx1250) — matrix operand order repaired and checked by reading, not compiled, not run on hardware
//
#include <hip/hip_runtime.h>
#include <math.h>

// ---------------- problem constants ----------------
#define PPTS   262144        // P
#define TSZ    524288        // 2^19 hash table entries per level
#define HMASK  (TSZ - 1)
#define LV     16
#define NPB    64            // points per block
#define THREADS 128          // 4 waves (wave32)

// LDS weight offsets (in _Float16 elements), transposed layout wT[n][kpad]
#define OFF_D0 0       // 64 x 32
#define OFF_D1 2048    // 16 x 64
#define OFF_C0 3072    // 64 x 64
#define OFF_C1 7168    // 64 x 64
#define OFF_C2 11264   // 16 x 64
#define OFF_S0 12288   // 64 x 32
#define OFF_S1 14336   // 64 x 64
#define OFF_S2 18432   // 16 x 64
#define WTOT   19456

// bias offsets (floats)
#define B_D0 0
#define B_D1 64
#define B_C0 80
#define B_C1 144
#define B_C2 208
#define B_S0 224
#define B_S1 288
#define B_S2 352
#define BTOT 368

typedef __attribute__((ext_vector_type(16))) _Float16 v16h;
typedef __attribute__((ext_vector_type(8)))  float    v8f;

union AFrag { v16h h; unsigned int u[8]; };
union BFrag { v16h h; uint4 q[2]; };

// NL[l] = floor(16 * (2048/16)^(l/15)) = floor(16 * 2^(7l/15))
__constant__ int NLTAB[LV] = {16, 22, 30, 42, 58, 80, 111, 153,
                              212, 294, 406, 561, 776, 1072, 1482, 2048};

// wave-local LDS fence: all prior ds stores visible to subsequent ds loads
__device__ inline void wave_lds_fence() {
    asm volatile("s_wait_dscnt 0" ::: "memory");
}

// Build A-fragment for V_WMMA_F32_16X16X32_F16 from an LDS activation tile.
// ISA layout (16-bit A 16x32): lanes 0-15 (M=lane): VGPR0-3 = K 0..7,
// VGPR4-7 = K 16..23; lanes 16-31: K 8..15 and K 24..31.
__device__ inline void loadA(AFrag* a, int kSteps, const _Float16* actIn,
                             int stride, int lane) {
    const int m  = lane & 15;
    const int hi = lane >> 4;
    const _Float16* row = actIn + m * stride;
    for (int ks = 0; ks < kSteps; ++ks) {
        const int kb = ks * 32 + hi * 8;
#pragma unroll
        for (int dj = 0; dj < 4; ++dj)
            a[ks].u[dj] = *reinterpret_cast<const unsigned int*>(row + kb + 2 * dj);
#pragma unroll
        for (int dj = 0; dj < 4; ++dj)
            a[ks].u[4 + dj] = *reinterpret_cast<const unsigned int*>(row + kb + 16 + 2 * dj);
    }
}

// One 16x16 output tile: D = A(16xK) * B(Kx16) + bias.
// B-fragment: lane n=lane&15 holds K = hi*8 .. hi*8+7 and 16+hi*8 .. 16+hi*8+7 (wT row n).
__device__ inline v8f wmma_tile(const AFrag* a, int kSteps, const _Float16* wT,
                                int kpad, int nglobal, float bv, int hi) {
    v8f acc = {bv, bv, bv, bv, bv, bv, bv, bv};
    for (int ks = 0; ks < kSteps; ++ks) {
        BFrag b;
        const uint4* bp =
            reinterpret_cast<const uint4*>(wT + nglobal * kpad + ks * 32 + hi * 8);
        b.q[0] = bp[0];
        b.q[1] = bp[2];
        acc = __builtin_amdgcn_wmma_f32_16x16x32_f16(
            /*neg_a=*/false, a[ks].h, /*neg_b=*/false, b.h,
            /*c_mod=*/(short)0, acc, /*reuse_a=*/false, /*reuse_b=*/false);
    }
    return acc;
}

// 64-wide hidden layer (4 N-tiles), ReLU, f16 output to LDS.
__device__ inline void dense_layer(const _Float16* actIn, int strideIn, int kSteps,
                                   const _Float16* wT, int kpad, const float* bias,
                                   _Float16* actOut, int strideOut, int lane) {
    AFrag a[2];
    loadA(a, kSteps, actIn, strideIn, lane);
    const int n = lane & 15, hi = lane >> 4;
#pragma unroll
    for (int nt = 0; nt < 4; ++nt) {
        const int ng = nt * 16 + n;
        v8f acc = wmma_tile(a, kSteps, wT, kpad, ng, bias[ng], hi);
#pragma unroll
        for (int r = 0; r < 8; ++r) {
            float v = acc[r];
            v = v > 0.f ? v : 0.f;
            actOut[(r + hi * 8) * strideOut + ng] = (_Float16)v;
        }
    }
}

// Stage w[K][N] (row-major f32) -> LDS transposed f16 wT[n][kpad], zero padded.
__device__ inline void stageW(const float* w, int K, int N, _Float16* dst,
                              int kpad, int npad, int tid) {
    const int tot = kpad * npad;
    for (int i = tid; i < tot; i += THREADS) {
        const int nn = i / kpad;
        const int k  = i - nn * kpad;
        float v = (k < K && nn < N) ? w[k * N + nn] : 0.0f;
        dst[nn * kpad + k] = (_Float16)v;
    }
}

__device__ inline void stageB(const float* b, int N, float* dst, int npad, int tid) {
    for (int i = tid; i < npad; i += THREADS) dst[i] = (i < N) ? b[i] : 0.0f;
}

__global__ __launch_bounds__(THREADS)
void ngp_fused(const float* __restrict__ x, const float* __restrict__ dvec,
               const float* __restrict__ tables,
               const float* dw0, const float* db0, const float* dw1, const float* db1,
               const float* cw0, const float* cb0, const float* cw1, const float* cb1,
               const float* cw2, const float* cb2,
               const float* sw0, const float* sb0, const float* sw1, const float* sb1,
               const float* sw2, const float* sb2,
               float* __restrict__ out) {
    __shared__ __align__(16) _Float16 s_w[WTOT];       // transposed f16 weights
    __shared__ float s_b[BTOT];                        // f32 biases
    __shared__ __align__(16) _Float16 s_actb[4][1024]; // per-wave act tile 16x64
    __shared__ __align__(16) _Float16 s_cib[4][1024];  // per-wave color input 16x64
    __shared__ float s_outb[4][256];                   // per-wave 16x16 f32 epilogue

    const int tid = threadIdx.x;

    // ---------- stage weights & biases ----------
    stageW(dw0, 32, 64, s_w + OFF_D0, 32, 64, tid);
    stageW(dw1, 64, 16, s_w + OFF_D1, 64, 16, tid);
    stageW(cw0, 43, 64, s_w + OFF_C0, 64, 64, tid);
    stageW(cw1, 64, 64, s_w + OFF_C1, 64, 64, tid);
    stageW(cw2, 64, 3,  s_w + OFF_C2, 64, 16, tid);
    stageW(sw0, 16, 64, s_w + OFF_S0, 32, 64, tid);
    stageW(sw1, 64, 64, s_w + OFF_S1, 64, 64, tid);
    stageW(sw2, 64, 10, s_w + OFF_S2, 64, 16, tid);
    stageB(db0, 64, s_b + B_D0, 64, tid);
    stageB(db1, 16, s_b + B_D1, 16, tid);
    stageB(cb0, 64, s_b + B_C0, 64, tid);
    stageB(cb1, 64, s_b + B_C1, 64, tid);
    stageB(cb2, 3,  s_b + B_C2, 16, tid);
    stageB(sb0, 64, s_b + B_S0, 64, tid);
    stageB(sb1, 64, s_b + B_S1, 64, tid);
    stageB(sb2, 10, s_b + B_S2, 16, tid);

    // ---------- hash-grid features + pos-enc (2 threads / point) ----------
    {
        const int p = tid >> 1, half = tid & 1;
        const int wv = p >> 4, m = p & 15;
        const size_t gp = (size_t)blockIdx.x * NPB + p;
        const float xs0 = x[gp * 3 + 0], xs1 = x[gp * 3 + 1], xs2 = x[gp * 3 + 2];
        const float x0 = xs0 + 0.5f, x1 = xs1 + 0.5f, x2 = xs2 + 0.5f;
        const float2* t2 = reinterpret_cast<const float2*>(tables);
        for (int li = 0; li < 8; ++li) {
            const int l = half * 8 + li;
            const float N = (float)NLTAB[l];
            const float a0 = x0 * N, a1 = x1 * N, a2 = x2 * N;
            const float fl0 = floorf(a0), fl1 = floorf(a1), fl2 = floorf(a2);
            const float r0 = a0 - fl0, r1 = a1 - fl1, r2 = a2 - fl2;
            const int vf0 = (int)fl0, vf1 = (int)fl1, vf2 = (int)fl2;
            const int vc0 = (int)ceilf(a0), vc1 = (int)ceilf(a1), vc2 = (int)ceilf(a2);
            float acc0 = 0.f, acc1 = 0.f;
#pragma unroll
            for (int c = 0; c < 8; ++c) {
                const int b0 = c & 1, b1 = (c >> 1) & 1, b2 = (c >> 2) & 1;
                const unsigned ix = (unsigned)(b0 ? vc0 : vf0);
                const unsigned iy = (unsigned)(b1 ? vc1 : vf1);
                const unsigned iz = (unsigned)(b2 ? vc2 : vf2);
                const unsigned hh =
                    (ix ^ (iy * 2654435761u) ^ (iz * 805459861u)) & HMASK;
                const float2 v = t2[(size_t)l * TSZ + hh];
                const float w = (b0 ? r0 : 1.f - r0) * (b1 ? r1 : 1.f - r1) *
                                (b2 ? r2 : 1.f - r2);
                acc0 += v.x * w;
                acc1 += v.y * w;
            }
            s_actb[wv][m * 64 + 2 * l]     = (_Float16)acc0;
            s_actb[wv][m * 64 + 2 * l + 1] = (_Float16)acc1;
        }
        if (half == 0) {
            const bool msk =
                (fabsf(xs0) < 0.5f) && (fabsf(xs1) < 0.5f) && (fabsf(xs2) < 0.5f);
            s_outb[wv][m * 16 + 14] = msk ? 1.0f : 0.0f;
            const float d0v = dvec[gp * 3 + 0], d1v = dvec[gp * 3 + 1],
                        d2v = dvec[gp * 3 + 2];
            _Float16* ci = &s_cib[wv][m * 64];
            ci[16] = (_Float16)d0v;
            ci[17] = (_Float16)d1v;
            ci[18] = (_Float16)d2v;
#pragma unroll
            for (int j = 0; j < 4; ++j) {
                const float sc = (float)(1 << j);
                ci[19 + 6 * j + 0] = (_Float16)sinf(sc * d0v);
                ci[19 + 6 * j + 1] = (_Float16)sinf(sc * d1v);
                ci[19 + 6 * j + 2] = (_Float16)sinf(sc * d2v);
                ci[22 + 6 * j + 0] = (_Float16)cosf(sc * d0v);
                ci[22 + 6 * j + 1] = (_Float16)cosf(sc * d1v);
                ci[22 + 6 * j + 2] = (_Float16)cosf(sc * d2v);
            }
            for (int c = 43; c < 64; ++c) ci[c] = (_Float16)0.f;
        }
    }
    __syncthreads();

    // ---------- fused MLPs: one 16-point tile per wave ----------
    const int lane = tid & 31;
    const int wv   = tid >> 5;
    const int n    = lane & 15, hi = lane >> 4;
    _Float16* ACT = s_actb[wv];
    _Float16* CI  = s_cib[wv];
    float*    OUT = s_outb[wv];

    // D0: 32 -> 64, relu
    dense_layer(ACT, 64, 1, s_w + OFF_D0, 32, s_b + B_D0, ACT, 64, lane);
    wave_lds_fence();
    // D1: 64 -> 16 (linear): h -> CI[0:16], ACT[0:16] (+zero pad 16:32), h0 -> OUT
    {
        AFrag a[2];
        loadA(a, 2, ACT, 64, lane);
        v8f hacc = wmma_tile(a, 2, s_w + OFF_D1, 64, n, s_b[B_D1 + n], hi);
#pragma unroll
        for (int r = 0; r < 8; ++r) {
            const int mr = r + hi * 8;
            const _Float16 hv = (_Float16)hacc[r];
            CI[mr * 64 + n]       = hv;
            ACT[mr * 64 + n]      = hv;
            ACT[mr * 64 + 16 + n] = (_Float16)0.f;
            if (n == 0) OUT[mr * 16 + 3] = hacc[r];  // h0 (f32) for sigma
        }
    }
    wave_lds_fence();
    // Segmentation branch: S0 16->64, S1 64->64, S2 64->10
    dense_layer(ACT, 64, 1, s_w + OFF_S0, 32, s_b + B_S0, ACT, 64, lane);
    wave_lds_fence();
    dense_layer(ACT, 64, 2, s_w + OFF_S1, 64, s_b + B_S1, ACT, 64, lane);
    wave_lds_fence();
    {
        AFrag a[2];
        loadA(a, 2, ACT, 64, lane);
        v8f sg = wmma_tile(a, 2, s_w + OFF_S2, 64, n, s_b[B_S2 + n], hi);
#pragma unroll
        for (int r = 0; r < 8; ++r) {
            const int mr = r + hi * 8;
            if (n < 10) OUT[mr * 16 + 4 + n] = sg[r];  // seg logits
        }
    }
    wave_lds_fence();
    // Color branch: C0 43->64, C1 64->64, C2 64->3
    dense_layer(CI, 64, 2, s_w + OFF_C0, 64, s_b + B_C0, ACT, 64, lane);
    wave_lds_fence();
    dense_layer(ACT, 64, 2, s_w + OFF_C1, 64, s_b + B_C1, ACT, 64, lane);
    wave_lds_fence();
    {
        AFrag a[2];
        loadA(a, 2, ACT, 64, lane);
        v8f cg = wmma_tile(a, 2, s_w + OFF_C2, 64, n, s_b[B_C2 + n], hi);
#pragma unroll
        for (int r = 0; r < 8; ++r) {
            const int mr = r + hi * 8;
            if (n < 3) OUT[mr * 16 + n] = cg[r];  // color logits
        }
    }
    __syncthreads();

    // ---------- epilogue: sigmoid / exp / softmax + mask, 1 thread / point ----------
    if (tid < NPB) {
        const int p = tid, w2 = p >> 4, m2 = p & 15;
        const float* o = &s_outb[w2][m2 * 16];
        const bool msk = o[14] > 0.5f;
        const size_t gp = (size_t)blockIdx.x * NPB + p;
#pragma unroll
        for (int c = 0; c < 3; ++c) {
            const float sv = 1.0f / (1.0f + expf(-o[c]));
            out[gp * 3 + c] = msk ? sv : 0.0f;
        }
        out[(size_t)3 * PPTS + gp] = msk ? expf(o[3]) : 0.0f;
        float mx = o[4];
#pragma unroll
        for (int j = 1; j < 10; ++j) mx = fmaxf(mx, o[4 + j]);
        float e[10];
        float sum = 0.f;
#pragma unroll
        for (int j = 0; j < 10; ++j) {
            e[j] = expf(o[4 + j] - mx);
            sum += e[j];
        }
        const float inv = 1.0f / sum;
#pragma unroll
        for (int j = 0; j < 10; ++j)
            out[(size_t)4 * PPTS + gp * 10 + j] = msk ? e[j] * inv : 0.0f;
    }
}

extern "C" void kernel_launch(void* const* d_in, const int* in_sizes, int n_in,
                              void* d_out, int out_size, void* d_ws, size_t ws_size,
                              hipStream_t stream) {
    (void)in_sizes; (void)n_in; (void)out_size; (void)d_ws; (void)ws_size;
    const float* x   = (const float*)d_in[0];
    const float* d   = (const float*)d_in[1];
    const float* tb  = (const float*)d_in[2];
    const float* dw0 = (const float*)d_in[3];
    const float* db0 = (const float*)d_in[4];
    const float* dw1 = (const float*)d_in[5];
    const float* db1 = (const float*)d_in[6];
    const float* cw0 = (const float*)d_in[7];
    const float* cb0 = (const float*)d_in[8];
    const float* cw1 = (const float*)d_in[9];
    const float* cb1 = (const float*)d_in[10];
    const float* cw2 = (const float*)d_in[11];
    const float* cb2 = (const float*)d_in[12];
    const float* sw0 = (const float*)d_in[13];
    const float* sb0 = (const float*)d_in[14];
    const float* sw1 = (const float*)d_in[15];
    const float* sb1 = (const float*)d_in[16];
    const float* sw2 = (const float*)d_in[17];
    const float* sb2 = (const float*)d_in[18];
    float* out = (float*)d_out;

    dim3 grid(PPTS / NPB), block(THREADS);
    hipLaunchKernelGGL(ngp_fused, grid, block, 0, stream,
                       x, d, tb, dw0, db0, dw1, db1, cw0, cb0, cw1, cb1, cw2, cb2,
                       sw0, sb0, sw1, sb1, sw2, sb2, out);
}
